// MambaModel_64630667870746
// MI455X (gfx1250) — matrix operand order repaired and checked by reading, not compiled, not run on hardware
//
#include <hip/hip_runtime.h>
#include <cstdint>
#include <cstddef>

// ============================================================================
// MI455X (gfx1250) forward pass for the VideoMamba U-Net reference.
// wave32, WMMA f32_16x16x32_f16 for every GEMM, Tensor Data Mover (TDM) for
// weight-tile staging, chunked parallel scan for the selective-state
// recurrence. fp32 storage, f16 WMMA inputs, f32 accumulate.
// ============================================================================

typedef _Float16 half_t;
typedef __attribute__((ext_vector_type(16))) _Float16 v16h;
typedef __attribute__((ext_vector_type(8)))  float    v8f;
typedef __attribute__((ext_vector_type(4)))  float    f4;
typedef __attribute__((ext_vector_type(4)))  _Float16 h4;
typedef __attribute__((ext_vector_type(4)))  unsigned int u32x4;
typedef __attribute__((ext_vector_type(8)))  int      i32x8;
typedef __attribute__((ext_vector_type(4)))  int      i32x4;

#if defined(__has_builtin)
#if __has_builtin(__builtin_amdgcn_tensor_load_to_lds) && __has_builtin(__builtin_amdgcn_s_wait_tensorcnt)
#define USE_TDM 1
#endif
#endif
#ifndef USE_TDM
#define USE_TDM 0
#endif

__device__ __forceinline__ float sigmoidf_(float x) { return 1.f / (1.f + __expf(-x)); }
__device__ __forceinline__ float softplusf_(float x) { return x > 20.f ? x : log1pf(__expf(x)); }

#define TM 128
#define TN 64
#define TK 32

#if USE_TDM
// Issue one TDM 2D tile load: gsrc (row stride ldw floats) -> LDS at lds_off.
// Tile = TN rows x TK cols of fp32; tensor dims are the REMAINING extents so
// hardware zero-fills out-of-bounds rows/cols. pad_enable inserts 1 dword of
// LDS padding after every 32 dwords => LDS row stride (TK+1) floats.
// D# bitfields per CDNA5 ISA.
__device__ __forceinline__ void tdm_load_w_tile(const float* gsrc, unsigned lds_off,
                                                int rem_k, int rem_n, int ldw)
{
  unsigned long long ga = (unsigned long long)(uintptr_t)gsrc;
  u32x4 g0;
  g0[0] = 1u;                                                // count=1 (valid), user mode
  g0[1] = lds_off;                                           // lds_addr [63:32]
  g0[2] = (unsigned)(ga & 0xffffffffu);                      // global_addr[31:0]
  g0[3] = (unsigned)((ga >> 32) & 0x1ffffffu) | (2u << 30);  // global_addr[56:32], type=2
  i32x8 g1;
  g1[0] = (2 << 16) | (1 << 20) | (4 << 22);   // data_size=4B, pad_enable, interval=32dw, pad=1dw
  g1[1] = (rem_k & 0xffff) << 16;              // tensor_dim0[15:0]   (bits 79:48)
  g1[2] = (rem_n & 0xffff) << 16;              // tensor_dim0[31:16]=0, tensor_dim1[15:0]
  g1[3] = (TK << 16);                          // tensor_dim1[31:16]=0, tile_dim0=TK
  g1[4] = TN;                                  // tile_dim1=TN, tile_dim2=0 (2D)
  g1[5] = ldw;                                 // tensor_dim0_stride[31:0]
  g1[6] = 0;                                   // stride[47:32] + dim1_stride lo (unused, 2D)
  g1[7] = 0;
  i32x4 z4 = {0, 0, 0, 0};
#if __clang_major__ >= 23
  i32x8 z8 = {};
  __builtin_amdgcn_tensor_load_to_lds(g0, g1, z4, z4, z8, 0);
#else
  __builtin_amdgcn_tensor_load_to_lds(g0, g1, z4, z4, 0);
#endif
}
#endif

// ---------------------------------------------------------------------------
// Generic GEMM:  C[M,N] = A[M,K] (row stride lda) * W[N,K]^T   (+ bias, +softplus)
// epi: 0 = none, 1 = +bias, 2 = +bias then softplus
// Tile 128x64/block, 8 waves, each wave a 32x32 tile = 4 WMMAs per K-step.
// A staged via float4 loads + f16 convert (interior) / clamped branchless (edge).
// W staged by the TDM, double-buffered, zero-filled at edges by hardware.
// ---------------------------------------------------------------------------
__launch_bounds__(256)
__global__ void k_gemm(const float* __restrict__ A, int lda,
                       const float* __restrict__ W,
                       const float* __restrict__ bias,
                       float* __restrict__ C,
                       int M, int N, int K, int epi)
{
  __shared__ half_t sA[TM][TK + 8];        // f16 A tile, rows 8B-aligned
#if USE_TDM
  __shared__ float sW[2][TN][TK + 1];      // fp32 W tiles; +1 = TDM pad dword
#else
  __shared__ half_t sB[TN][TK + 8];
#endif
  const int tid  = threadIdx.x;
  const int wave = tid >> 5;
  const int lane = tid & 31;
  const int wm   = wave & 3;    // 4 waves along M
  const int wn   = wave >> 2;   // 2 waves along N
  const int m0   = blockIdx.x * TM;
  const int n0   = blockIdx.y * TN;

  const int arow = tid >> 1, akk = (tid & 1) * 16;   // A: 16 halves/thread
  const int agm  = m0 + arow;
  const bool m_full = (m0 + TM <= M);                // block-uniform

#if !USE_TDM
  const int brow = tid >> 2, bkk = (tid & 3) * 8;
  const int bgn  = n0 + brow;
  const bool n_full = (n0 + TN <= N);
#endif

  v8f acc00 = {}, acc01 = {}, acc10 = {}, acc11 = {};

#if USE_TDM
  if (wave == 0) {   // wave-uniform branch; TDM ignores EXEC, issues once per wave
    tdm_load_w_tile(W + (size_t)n0 * K, (unsigned)(uintptr_t)&sW[0][0][0],
                    K, N - n0, K);
  }
  int buf = 0;
#endif

  for (int k0 = 0; k0 < K; k0 += TK) {
    // ---- stage A ----
    if (m_full && (k0 + TK <= K)) {
      const float* pa = A + (size_t)agm * lda + k0 + akk;
      __builtin_prefetch(pa + TK, 0, 1);   // global_prefetch for next K tile
#pragma unroll
      for (int v = 0; v < 4; ++v) {
        f4 f = *(const f4*)(pa + 4 * v);
        h4 h = { (half_t)f.x, (half_t)f.y, (half_t)f.z, (half_t)f.w };
        *(h4*)&sA[arow][akk + 4 * v] = h;
      }
    } else {
      const int cgm = (agm < M) ? agm : (M - 1);
#pragma unroll
      for (int i = 0; i < 16; ++i) {
        const int gk  = k0 + akk + i;
        const int cgk = (gk < K) ? gk : (K - 1);
        float v = A[(size_t)cgm * lda + cgk];
        sA[arow][akk + i] = (half_t)((agm < M && gk < K) ? v : 0.f);
      }
    }

#if USE_TDM
    // ---- prefetch next W tile via TDM, ensure current tile landed ----
    if (wave == 0) {
      if (k0 + TK < K) {
        tdm_load_w_tile(W + (size_t)n0 * K + (k0 + TK),
                        (unsigned)(uintptr_t)&sW[buf ^ 1][0][0],
                        K - (k0 + TK), N - n0, K);
        __builtin_amdgcn_s_wait_tensorcnt(1);   // current tile complete
      } else {
        __builtin_amdgcn_s_wait_tensorcnt(0);
      }
    }
#else
    if (n_full && (k0 + TK <= K)) {
      const float* pb = W + (size_t)bgn * K + k0 + bkk;
#pragma unroll
      for (int v = 0; v < 2; ++v) {
        f4 f = *(const f4*)(pb + 4 * v);
        h4 h = { (half_t)f.x, (half_t)f.y, (half_t)f.z, (half_t)f.w };
        *(h4*)&sB[brow][bkk + 4 * v] = h;
      }
    } else {
      const int cgn = (bgn < N) ? bgn : (N - 1);
#pragma unroll
      for (int i = 0; i < 8; ++i) {
        const int gk  = k0 + bkk + i;
        const int cgk = (gk < K) ? gk : (K - 1);
        float v = W[(size_t)cgn * K + cgk];
        sB[brow][bkk + i] = (half_t)((bgn < N && gk < K) ? v : 0.f);
      }
    }
#endif
    __syncthreads();

    // Fragment loads in contiguous K order (this code's choice; A and B match):
    // A 16x32: lane l (group g=l>>4) holds row l&15, K = 16g + i
    // B 32x16: lane l holds col l&15, K = 16g + i
    const int fm   = lane & 15;
    const int ko8  = (lane >> 4) * 8;
    const int ko16 = (lane >> 4) * 16;
    v16h a0, a1, b0, b1;
#pragma unroll
    for (int i = 0; i < 8; ++i) {
      a0[i]     = sA[wm * 32 + fm][ko16 + i];
      a0[8 + i] = sA[wm * 32 + fm][ko16 + 8 + i];
      a1[i]     = sA[wm * 32 + 16 + fm][ko16 + i];
      a1[8 + i] = sA[wm * 32 + 16 + fm][ko16 + 8 + i];
    }
#if USE_TDM
#pragma unroll
    for (int i = 0; i < 16; ++i) {
      b0[i] = (half_t)sW[buf][wn * 32 + fm][ko16 + i];
      b1[i] = (half_t)sW[buf][wn * 32 + 16 + fm][ko16 + i];
    }
#else
#pragma unroll
    for (int i = 0; i < 16; ++i) {
      b0[i] = sB[wn * 32 + fm][ko16 + i];
      b1[i] = sB[wn * 32 + 16 + fm][ko16 + i];
    }
#endif
    // EXEC is all-ones here (reconverged before WMMA).
    acc00 = __builtin_amdgcn_wmma_f32_16x16x32_f16(false, a0, false, b0, (short)0, acc00, false, false);
    acc01 = __builtin_amdgcn_wmma_f32_16x16x32_f16(false, a0, false, b1, (short)0, acc01, false, false);
    acc10 = __builtin_amdgcn_wmma_f32_16x16x32_f16(false, a1, false, b0, (short)0, acc10, false, false);
    acc11 = __builtin_amdgcn_wmma_f32_16x16x32_f16(false, a1, false, b1, (short)0, acc11, false, false);
    __syncthreads();
#if USE_TDM
    buf ^= 1;
#endif
  }

  // C/D layout: VGPR j -> row 8*(lane>=16)+j, col = lane&15
  const int cn = lane & 15;
  const int r8 = (lane >> 4) * 8;
#pragma unroll
  for (int t = 0; t < 4; ++t) {
    const int mt = (t >> 1) * 16;
    const int nt = (t & 1) * 16;
    v8f acc = (t == 0) ? acc00 : (t == 1) ? acc01 : (t == 2) ? acc10 : acc11;
    const int col = n0 + wn * 32 + nt + cn;
    if (col < N) {
      const float bv = (epi >= 1) ? bias[col] : 0.f;
#pragma unroll
      for (int j = 0; j < 8; ++j) {
        const int row = m0 + wm * 32 + mt + r8 + j;
        if (row < M) {
          float v = acc[j] + bv;
          if (epi == 2) v = softplusf_(v);
          C[(size_t)row * N + col] = v;
        }
      }
    }
  }
}

// ---------------------------------------------------------------------------
// Patch embedding: x[1,1,3,360,640] conv 5x5 stride 5 -> out[9216,96]
// ---------------------------------------------------------------------------
__global__ void k_patch(const float* __restrict__ x, const float* __restrict__ pw,
                        const float* __restrict__ pb, float* __restrict__ out)
{
  int idx = blockIdx.x * blockDim.x + threadIdx.x;
  if (idx >= 9216 * 96) return;
  int e = idx % 96, tok = idx / 96;
  int h = tok / 128, w = tok % 128;
  float acc = pb[e];
#pragma unroll
  for (int c = 0; c < 3; ++c)
#pragma unroll
    for (int p = 0; p < 5; ++p)
#pragma unroll
      for (int q = 0; q < 5; ++q)
        acc += x[(size_t)(c * 360 + h * 5 + p) * 640 + w * 5 + q] *
               pw[((e * 3 + c) * 5 + p) * 5 + q];
  out[idx] = acc;
}

// cls + pos_embed + temporal_pos assembly -> [9217,96]
__global__ void k_assemble(const float* __restrict__ pe, const float* __restrict__ cls,
                           const float* __restrict__ pos, const float* __restrict__ tpos,
                           float* __restrict__ out)
{
  int idx = blockIdx.x * blockDim.x + threadIdx.x;
  if (idx >= 9217 * 96) return;
  int e = idx % 96, t = idx / 96;
  float v = pos[idx];
  v += (t == 0) ? cls[e] : (pe[(size_t)(t - 1) * 96 + e] + tpos[e]);
  out[idx] = v;
}

// Depthwise causal conv width-4 (+bias) + SiLU.  xi = xz[:, 0:di]
__global__ void k_conv(const float* __restrict__ xz, const float* __restrict__ cw,
                       const float* __restrict__ cb, float* __restrict__ xc, int L, int di)
{
  int idx = blockIdx.x * blockDim.x + threadIdx.x;
  if (idx >= L * di) return;
  int d = idx % di, l = idx / di;
  int s2 = 2 * di;
  float acc = cb[d];
#pragma unroll
  for (int j = 0; j < 4; ++j) {
    int ls = l - 3 + j;
    if (ls >= 0) acc += xz[(size_t)ls * s2 + d] * cw[d * 4 + j];
  }
  xc[idx] = acc * sigmoidf_(acc);
}

// ---------------------------------------------------------------------------
// Selective scan: h[l] = exp(delta*A)*h[l-1] + delta*B*x, chunked 3-pass.
// ---------------------------------------------------------------------------
__global__ void k_scan1(const float* __restrict__ delta, const float* __restrict__ xc,
                        const float* __restrict__ dbc, const float* __restrict__ A_log,
                        float* __restrict__ aggA, float* __restrict__ aggH,
                        int L, int di, int dbcC, int dtr, int CH)
{
  int d = blockIdx.y * blockDim.x + threadIdx.x;
  if (d >= di) return;
  int c = blockIdx.x;
  float a[16], ap[16], h[16];
#pragma unroll
  for (int s = 0; s < 16; ++s) { a[s] = -__expf(A_log[d * 16 + s]); ap[s] = 1.f; h[s] = 0.f; }
  int l0 = c * CH, l1 = l0 + CH; if (l1 > L) l1 = L;
  for (int l = l0; l < l1; ++l) {
    float dl = delta[(size_t)l * di + d];
    float bx = dl * xc[(size_t)l * di + d];
    const float* Bm = dbc + (size_t)l * dbcC + dtr;
#pragma unroll
    for (int s = 0; s < 16; ++s) {
      float dA = __expf(dl * a[s]);
      h[s] = dA * h[s] + bx * Bm[s];
      ap[s] *= dA;
    }
  }
  size_t o = ((size_t)c * di + d) * 16;
#pragma unroll
  for (int s = 0; s < 16; ++s) { aggA[o + s] = ap[s]; aggH[o + s] = h[s]; }
}

__global__ void k_scan2(const float* __restrict__ aggA, const float* __restrict__ aggH,
                        float* __restrict__ carry, int di, int NC)
{
  int t = blockIdx.x * blockDim.x + threadIdx.x;   // t = d*16 + s
  if (t >= di * 16) return;
  float cr = 0.f;
  for (int c = 0; c < NC; ++c) {
    size_t o = (size_t)c * di * 16 + t;
    carry[o] = cr;                       // exclusive prefix (state before chunk c)
    cr = aggA[o] * cr + aggH[o];         // first-order recurrence composition
  }
}

// pass 3: replay with carry + fused epilogue  y = (sum_s h*C + xc*D) * silu(z)
__global__ void k_scan3(const float* __restrict__ delta, const float* __restrict__ xc,
                        const float* __restrict__ dbc, const float* __restrict__ A_log,
                        const float* __restrict__ carry, const float* __restrict__ Dp,
                        const float* __restrict__ xz, float* __restrict__ y,
                        int L, int di, int dbcC, int dtr, int CH)
{
  int d = blockIdx.y * blockDim.x + threadIdx.x;
  if (d >= di) return;
  int c = blockIdx.x;
  float a[16], h[16];
  size_t o = ((size_t)c * di + d) * 16;
#pragma unroll
  for (int s = 0; s < 16; ++s) { a[s] = -__expf(A_log[d * 16 + s]); h[s] = carry[o + s]; }
  float Dv = Dp[d];
  int l0 = c * CH, l1 = l0 + CH; if (l1 > L) l1 = L;
  for (int l = l0; l < l1; ++l) {
    float dl  = delta[(size_t)l * di + d];
    float xcl = xc[(size_t)l * di + d];
    float bx  = dl * xcl;
    const float* Bm = dbc + (size_t)l * dbcC + dtr;
    const float* Cm = Bm + 16;
    float acc = 0.f;
#pragma unroll
    for (int s = 0; s < 16; ++s) {
      float dA = __expf(dl * a[s]);
      h[s] = dA * h[s] + bx * Bm[s];
      acc += h[s] * Cm[s];
    }
    float zv = xz[(size_t)l * 2 * di + di + d];
    y[(size_t)l * di + d] = (acc + xcl * Dv) * (zv * sigmoidf_(zv));
  }
}

// ---------------------------------------------------------------------------
// BatchNorm over tokens: per-feature stats -> scale/shift -> apply
// ---------------------------------------------------------------------------
__global__ void k_bn_stats(const float* __restrict__ x, const float* __restrict__ w,
                           const float* __restrict__ b, float* __restrict__ scale,
                           float* __restrict__ shift, int L, int d)
{
  int f = blockIdx.x * blockDim.x + threadIdx.x;
  if (f >= d) return;
  float s = 0.f, sq = 0.f;
  for (int l = 0; l < L; ++l) { float v = x[(size_t)l * d + f]; s += v; sq += v * v; }
  float mu = s / L, var = sq / L - mu * mu;
  float sc = w[f] * rsqrtf(var + 1e-5f);
  scale[f] = sc; shift[f] = b[f] - mu * sc;
}

__global__ void k_bn_apply(float* __restrict__ x, const float* __restrict__ scale,
                           const float* __restrict__ shift, int n, int d)
{
  int idx = blockIdx.x * blockDim.x + threadIdx.x;
  if (idx >= n) return;
  int f = idx % d;
  x[idx] = x[idx] * scale[f] + shift[f];
}

// LayerNorm over last dim C, one wave per token (wave32), in-place.
__global__ void k_ln(float* __restrict__ x, const float* __restrict__ w,
                     const float* __restrict__ b, int ntok, int C)
{
  int wave = threadIdx.x >> 5, lane = threadIdx.x & 31;
  int t = blockIdx.x * 8 + wave;
  if (t >= ntok) return;
  float* row = x + (size_t)t * C;
  float s = 0.f, sq = 0.f;
  for (int f = lane; f < C; f += 32) { float v = row[f]; s += v; sq += v * v; }
#pragma unroll
  for (int m = 16; m; m >>= 1) { s += __shfl_xor(s, m, 32); sq += __shfl_xor(sq, m, 32); }
  float mu = s / C, var = sq / C - mu * mu;
  float rs = rsqrtf(var + 1e-5f);
  for (int f = lane; f < C; f += 32) row[f] = (row[f] - mu) * rs * w[f] + b[f];
}

// Encoder 2x2 patch-merge gather: body of xin [1+Hn*Wn, C] -> out4 [N2, 4C]
__global__ void k_merge(const float* __restrict__ xin, float* __restrict__ out4,
                        int Hn, int Wn, int C)
{
  int Wh = Wn >> 1;
  int C4 = 4 * C;
  int N2 = (Hn >> 1) * Wh;
  int idx = blockIdx.x * blockDim.x + threadIdx.x;
  if (idx >= N2 * C4) return;
  int n2 = idx / C4, j = idx % C4;
  int q = j / C, ch = j % C;
  int r = q & 1, cc = q >> 1;
  int h = n2 / Wh, w = n2 % Wh;
  int src = (2 * h + r) * Wn + (2 * w + cc);
  out4[idx] = xin[(size_t)(1 + src) * C + ch];
}

// Decoder expand scatter: tin [4N, m] (token order n,r,c) -> body of xout [1+4N, m]
__global__ void k_scatter(const float* __restrict__ tin, float* __restrict__ xout,
                          int Hn, int Wn, int m)
{
  int N4 = 4 * Hn * Wn;
  int idx = blockIdx.x * blockDim.x + threadIdx.x;
  if (idx >= N4 * m) return;
  int tk = idx / m, i = idx % m;
  int n = tk >> 2, rc = tk & 3;
  int r = rc >> 1, cc = rc & 1;
  int h = n / Wn, w = n % Wn;
  int t = (2 * h + r) * (2 * Wn) + (2 * w + cc);
  xout[(size_t)(1 + t) * m + i] = tin[idx];
}

__global__ void k_add(float* __restrict__ x, const float* __restrict__ s, int n)
{
  int idx = blockIdx.x * blockDim.x + threadIdx.x;
  if (idx < n) x[idx] += s[idx];
}

// ============================================================================
// Host orchestration
// ============================================================================
struct MambaP {
  const float *in_w, *conv_w, *conv_b, *xproj_w, *dt_w, *dt_b, *A_log, *D, *out_w;
  int d, di, dtr;
};
struct MBlockP { MambaP m0, m1; const float *bn_w, *bn_b; int d; };
struct EncP { MBlockP mb; const float *norm_w, *norm_b, *red_w, *cls_w, *cls_b; };
struct DecP { MBlockP mb; const float *exp_w, *norm_w, *norm_b, *cls_w, *cls_b; };

struct Scratch {
  float *xz, *xc, *dbc, *delta, *ybuf, *aggA, *aggH, *carry, *bnsc, *bnsh, *tbuf;
};

static inline void gemm(hipStream_t st, const float* A, int lda, const float* W,
                        const float* bias, float* C, int M, int N, int K, int epi)
{
  dim3 g((M + TM - 1) / TM, (N + TN - 1) / TN);
  k_gemm<<<g, 256, 0, st>>>(A, lda, W, bias, C, M, N, K, epi);
}

static void mamba_layer(hipStream_t st, const MambaP& m, float* x, int L, const Scratch& S)
{
  const int d = m.d, di = m.di, dtr = m.dtr;
  const int dbcC = dtr + 32;
  gemm(st, x, d, m.in_w, nullptr, S.xz, L, 2 * di, d, 0);
  k_conv<<<((L * di) + 255) / 256, 256, 0, st>>>(S.xz, m.conv_w, m.conv_b, S.xc, L, di);
  gemm(st, S.xc, di, m.xproj_w, nullptr, S.dbc, L, dbcC, di, 0);
  gemm(st, S.dbc, dbcC, m.dt_w, m.dt_b, S.delta, L, di, dtr, 2);
  const int CH = 64;
  const int NC = (L + CH - 1) / CH;
  dim3 g1(NC, (di + 255) / 256);
  k_scan1<<<g1, 256, 0, st>>>(S.delta, S.xc, S.dbc, m.A_log, S.aggA, S.aggH, L, di, dbcC, dtr, CH);
  k_scan2<<<((di * 16) + 255) / 256, 256, 0, st>>>(S.aggA, S.aggH, S.carry, di, NC);
  k_scan3<<<g1, 256, 0, st>>>(S.delta, S.xc, S.dbc, m.A_log, S.carry, m.D, S.xz, S.ybuf,
                              L, di, dbcC, dtr, CH);
  gemm(st, S.ybuf, di, m.out_w, nullptr, x, L, d, di, 0);
}

static void mblock(hipStream_t st, const MBlockP& b, float* x, int L, const Scratch& S)
{
  mamba_layer(st, b.m0, x, L, S);
  mamba_layer(st, b.m1, x, L, S);
  const int d = b.d;
  k_bn_stats<<<(d + 255) / 256, 256, 0, st>>>(x, b.bn_w, b.bn_b, S.bnsc, S.bnsh, L, d);
  k_bn_apply<<<((L * d) + 255) / 256, 256, 0, st>>>(x, S.bnsc, S.bnsh, L * d, d);
}

static void encoder(hipStream_t st, const EncP& e, const float* xin, float* xout,
                    int Hn, int Wn, int C, const Scratch& S)
{
  const int N2 = (Hn / 2) * (Wn / 2);
  const int C4 = 4 * C, C2 = 2 * C;
  k_merge<<<((N2 * C4) + 255) / 256, 256, 0, st>>>(xin, S.tbuf, Hn, Wn, C);
  k_ln<<<(N2 + 7) / 8, 256, 0, st>>>(S.tbuf, e.norm_w, e.norm_b, N2, C4);
  gemm(st, S.tbuf, C4, e.red_w, nullptr, xout + C2, N2, C2, C4, 0);   // body rows 1..
  gemm(st, xin, C, e.cls_w, e.cls_b, xout, 1, C2, C, 1);              // cls row 0
  mblock(st, e.mb, xout, 1 + N2, S);
}

static void decoder(hipStream_t st, const DecP& dd, const float* xin, float* xout,
                    int Hn, int Wn, int dim, const float* skip, const Scratch& S)
{
  const int N = Hn * Wn, m = dim / 2, N4 = 4 * N;
  gemm(st, xin + dim, dim, dd.exp_w, nullptr, S.tbuf, N, 2 * dim, dim, 0);
  k_ln<<<(N4 + 7) / 8, 256, 0, st>>>(S.tbuf, dd.norm_w, dd.norm_b, N4, m);
  k_scatter<<<((N4 * m) + 255) / 256, 256, 0, st>>>(S.tbuf, xout, Hn, Wn, m);
  gemm(st, xin, dim, dd.cls_w, dd.cls_b, xout, 1, m, dim, 1);
  k_add<<<(((1 + N4) * m) + 255) / 256, 256, 0, st>>>(xout, skip, (1 + N4) * m);
  mblock(st, dd.mb, xout, 1 + N4, S);
}

extern "C" void kernel_launch(void* const* d_in, const int* in_sizes, int n_in,
                              void* d_out, int out_size, void* d_ws, size_t ws_size,
                              hipStream_t stream)
{
  (void)in_sizes; (void)n_in; (void)out_size; (void)ws_size;
  // --- walk inputs in setup_inputs() dict insertion order (recursive) ---
  int c = 0;
  auto F = [&](int i) { return (const float*)d_in[i]; };
  const float* x_img   = F(c++);                    // x [1,1,3,360,640]
  const float* patch_w = F(c++);
  const float* patch_b = F(c++);
  const float* cls_tok = F(c++);
  const float* pos_emb = F(c++);
  const float* tpos    = F(c++);
  auto read_mamba = [&](MambaP& m, int d) {
    m.d = d; m.di = 2 * d; m.dtr = (d + 15) / 16;
    m.in_w = F(c++); m.conv_w = F(c++); m.conv_b = F(c++); m.xproj_w = F(c++);
    m.dt_w = F(c++); m.dt_b = F(c++); m.A_log = F(c++); m.D = F(c++); m.out_w = F(c++);
  };
  auto read_mblock = [&](MBlockP& b, int d) {
    b.d = d; read_mamba(b.m0, d); read_mamba(b.m1, d);
    b.bn_w = F(c++); b.bn_b = F(c++);
  };
  auto read_enc = [&](EncP& e, int dim) {
    read_mblock(e.mb, 2 * dim);
    e.norm_w = F(c++); e.norm_b = F(c++); e.red_w = F(c++); e.cls_w = F(c++); e.cls_b = F(c++);
  };
  auto read_dec = [&](DecP& d2, int dim) {
    read_mblock(d2.mb, dim / 2);
    d2.exp_w = F(c++); d2.norm_w = F(c++); d2.norm_b = F(c++); d2.cls_w = F(c++); d2.cls_b = F(c++);
  };
  MBlockP block1; read_mblock(block1, 96);
  EncP block2; read_enc(block2, 96);
  EncP block3; read_enc(block3, 192);
  EncP bott;   read_enc(bott, 384);
  DecP block4; read_dec(block4, 768);
  DecP block5; read_dec(block5, 384);
  DecP block6; read_dec(block6, 192);

  // --- workspace bump allocator (max over all stages) ---
  char* wp = (char*)d_ws;
  auto alloc = [&](size_t nfloats) {
    float* p = (float*)wp;
    wp += ((nfloats * sizeof(float) + 255) / 256) * 256;
    return p;
  };
  Scratch S;
  S.xz    = alloc(3539328);   // max L*2di = 9217*384
  S.xc    = alloc(1769664);   // max L*di  = 9217*192
  S.dbc   = alloc(737360);
  S.delta = alloc(1769664);
  S.ybuf  = alloc(1769664);
  S.aggA  = alloc(524288);    // max NC*di*16 = 145*192*16
  S.aggH  = alloc(524288);
  S.carry = alloc(524288);
  S.bnsc  = alloc(2048);
  S.bnsh  = alloc(2048);
  S.tbuf  = alloc(900000);
  float* bufA  = alloc(900000);
  float* bufB  = alloc(900000);
  float* skip1 = alloc(884832);  // 9217*96
  float* skip2 = alloc(442560);  // 2305*192
  float* skip3 = alloc(221568);  // 577*384

  // --- stem ---
  k_patch<<<(9216 * 96 + 255) / 256, 256, 0, stream>>>(x_img, patch_w, patch_b, S.tbuf);
  k_assemble<<<(9217 * 96 + 255) / 256, 256, 0, stream>>>(S.tbuf, cls_tok, pos_emb, tpos, bufA);

  // --- block1 ---
  mblock(stream, block1, bufA, 9217, S);
  hipMemcpyAsync(skip1, bufA, (size_t)9217 * 96 * sizeof(float), hipMemcpyDeviceToDevice, stream);

  // --- encoders ---
  encoder(stream, block2, bufA, bufB, 72, 128, 96, S);    // -> [2305,192]
  hipMemcpyAsync(skip2, bufB, (size_t)2305 * 192 * sizeof(float), hipMemcpyDeviceToDevice, stream);
  encoder(stream, block3, bufB, bufA, 36, 64, 192, S);    // -> [577,384]
  hipMemcpyAsync(skip3, bufA, (size_t)577 * 384 * sizeof(float), hipMemcpyDeviceToDevice, stream);
  encoder(stream, bott, bufA, bufB, 18, 32, 384, S);      // -> [145,768]

  // --- decoders ---
  decoder(stream, block4, bufB, bufA, 9, 16, 768, skip3, S);    // -> [577,384]
  decoder(stream, block5, bufA, bufB, 18, 32, 384, skip2, S);   // -> [2305,192]
  decoder(stream, block6, bufB, bufA, 36, 64, 192, skip1, S);   // -> [9217,96]

  hipMemcpyAsync(d_out, bufA, (size_t)9217 * 96 * sizeof(float), hipMemcpyDeviceToDevice, stream);
}
